// MalariaGAT_20693152432481
// MI455X (gfx1250) — matrix operand order repaired and checked by reading, not compiled, not run on hardware
//
#include <hip/hip_runtime.h>
#include <hip/hip_bf16.h>

typedef __attribute__((ext_vector_type(16))) _Float16 v16h;
typedef __attribute__((ext_vector_type(8)))  _Float16 v8h;
typedef __attribute__((ext_vector_type(8)))  float    v8f;

#define HGAT 4
#define CGAT 64
#define HCG  256
#define EDG  4

static __device__ __forceinline__ void atomicMaxF(float* addr, float value) {
    if (value >= 0.f) atomicMax((int*)addr, __float_as_int(value));
    else              atomicMin((unsigned int*)addr, __float_as_uint(value));
}

// ---------------- utility kernels ----------------

__global__ void fill_f32(float* __restrict__ p, float v, int count) {
    int t = blockIdx.x * 256 + threadIdx.x;
    if (t < count) p[t] = v;
}

__global__ void conv_f16(const float* __restrict__ src, _Float16* __restrict__ dst, int count) {
    int t = blockIdx.x * 256 + threadIdx.x;
    if (t < count) dst[t] = (_Float16)src[t];
}

// Pack row-major f32 B[K,Nc] into WMMA B-fragment order (f16).
// Fragment for (tileN tn, kblock kb): 32 lanes x 16 halves contiguous.
// lane<16 : N=lane,    K = kb*32 + 2v + (v>=4 ? 8 : 0), +1
// lane>=16: N=lane-16, K = kb*32 + 8 + 2v + (v>=4 ? 8 : 0), +1
__global__ void pack_b(const float* __restrict__ B, _Float16* __restrict__ Bp, int K, int Nc) {
    int tid = blockIdx.x * 256 + threadIdx.x;
    int KB = K >> 5;
    int total = (Nc >> 4) * KB * 32 * 8;   // dwords (pairs of halves)
    if (tid >= total) return;
    int v    = tid & 7;
    int lane = (tid >> 3) & 31;
    int rest = tid >> 8;
    int kb   = rest % KB;
    int tn   = rest / KB;
    int n = tn * 16 + (lane & 15);
    int k = kb * 32 + 8 * (lane >> 4) + 2 * v + ((v >> 2) << 3);
    Bp[(size_t)tid * 2]     = (_Float16)B[(size_t)k * Nc + n];
    Bp[(size_t)tid * 2 + 1] = (_Float16)B[(size_t)(k + 1) * Nc + n];
}

// Me[j*H+h] = sum_c We[j, h*C + c] * a_e[h*C + c]   (ED x H = 16 values)
__global__ void compute_me(const float* __restrict__ We, const float* __restrict__ ae,
                           float* __restrict__ Me) {
    int t = threadIdx.x;
    if (t >= EDG * HGAT) return;
    int j = t / HGAT, h = t % HGAT;
    float s = 0.f;
    for (int c = 0; c < CGAT; ++c)
        s += We[(size_t)j * HCG + h * CGAT + c] * ae[h * CGAT + c];
    Me[t] = s;
}

// ---------------- WMMA GEMM: C[M,Nc] = A[M,K](f16) @ Bpacked + bias ----------------

__global__ __launch_bounds__(128)
void gemm_wmma_f16(const _Float16* __restrict__ A, const _Float16* __restrict__ Bp,
                   float* __restrict__ C, const float* __restrict__ bias,
                   int M, int K, int Nc) {
    const int lane  = threadIdx.x & 31;
    const int wave  = threadIdx.x >> 5;
    const int tilesN = Nc >> 4;
    const int KB     = K >> 5;
    const int tileId = blockIdx.x * 4 + wave;
    if (tileId >= (M >> 4) * tilesN) return;   // wave-uniform exit (EXEC stays all-1)
    const int tm = tileId / tilesN;
    const int tn = tileId - tm * tilesN;

    const int half = lane >> 4;
    const _Float16* arow = A + (size_t)(tm * 16 + (lane & 15)) * K + half * 8;
    const _Float16* bptr = Bp + ((size_t)tn * KB * 32 + lane) * 16;

    v8f acc = {};
    for (int kb = 0; kb < KB; ++kb) {
        v16h a, b;
        *(v8h*)&a         = *(const v8h*)(arow);        // K = half*8 .. +7
        *(((v8h*)&a) + 1) = *(const v8h*)(arow + 16);   // K = 16 + half*8 .. +7
        b = *(const v16h*)(bptr);
        acc = __builtin_amdgcn_wmma_f32_16x16x32_f16(false, a, false, b,
                                                     (short)0, acc, false, false);
        arow += 32;
        bptr += 512;
    }
    const int col  = tn * 16 + (lane & 15);
    const int row0 = tm * 16 + half * 8;
    float bv = bias ? bias[col] : 0.f;
#pragma unroll
    for (int r = 0; r < 8; ++r)
        C[(size_t)(row0 + r) * Nc + col] = acc[r] + bv;
}

// ---------------- self-loop mean attr ----------------

__global__ void edge_mean_acc(const int* __restrict__ ei, const float* __restrict__ eattr,
                              float* __restrict__ sums, float* __restrict__ cnt, int Eo) {
    int e = blockIdx.x * 256 + threadIdx.x;
    if (e >= Eo) return;
    int d = ei[Eo + e];
    atomicAdd(&cnt[d], 1.f);
#pragma unroll
    for (int j = 0; j < EDG; ++j)
        atomicAdd(&sums[d * EDG + j], eattr[(size_t)e * EDG + j]);
}

__global__ void compute_mean(const float* __restrict__ sums, const float* __restrict__ cnt,
                             float* __restrict__ meanA, int Nn) {
    int t = blockIdx.x * 256 + threadIdx.x;
    if (t >= Nn * EDG) return;
    meanA[t] = sums[t] / fmaxf(cnt[t >> 2], 1.f);
}

// ---------------- attention logits per node ----------------

__global__ void alpha_node(const float* __restrict__ XS, const float* __restrict__ att_src,
                           const float* __restrict__ att_dst,
                           float* __restrict__ asrc, float* __restrict__ adst, int Nn) {
    int t = blockIdx.x * 256 + threadIdx.x;     // t = n*H + h
    if (t >= Nn * HGAT) return;
    int n = t >> 2, h = t & 3;
    const float* row = XS + (size_t)n * HCG + h * CGAT;
    float s = 0.f, d = 0.f;
    for (int c = 0; c < CGAT; ++c) {
        float v = row[c];
        s += v * att_src[h * CGAT + c];
        d += v * att_dst[h * CGAT + c];
    }
    asrc[t] = s;
    adst[t] = d;
}

// ---------------- edge passes ----------------

__global__ void edge_alpha_max(const int* __restrict__ ei, const float* __restrict__ eattr,
                               const float* __restrict__ meanA, const float* __restrict__ Me,
                               const float* __restrict__ asrc, const float* __restrict__ adst,
                               float* __restrict__ alphaE, float* __restrict__ mbuf,
                               int Eo, int Nn) {
    int t = blockIdx.x * 256 + threadIdx.x;     // t = e*H + h
    int e = t >> 2, h = t & 3;
    if (e >= Eo + Nn) return;
    int s, d;
    const float* ap;
    if (e < Eo) { s = ei[e]; d = ei[Eo + e]; ap = eattr + (size_t)e * EDG; }
    else        { s = e - Eo; d = s;         ap = meanA + (size_t)(e - Eo) * EDG; }
    float ae = ap[0] * Me[h] + ap[1] * Me[4 + h] + ap[2] * Me[8 + h] + ap[3] * Me[12 + h];
    float a = asrc[s * HGAT + h] + adst[d * HGAT + h] + ae;
    a = (a > 0.f) ? a : 0.2f * a;               // leaky relu, slope 0.2
    alphaE[t] = a;
    atomicMaxF(&mbuf[d * HGAT + h], a);
}

__global__ void edge_exp_denom(const int* __restrict__ ei, float* __restrict__ alphaE,
                               const float* __restrict__ mbuf, float* __restrict__ denom,
                               int Eo, int Nn) {
    int t = blockIdx.x * 256 + threadIdx.x;
    int e = t >> 2, h = t & 3;
    if (e >= Eo + Nn) return;
    int d = (e < Eo) ? ei[Eo + e] : (e - Eo);
    float p = __expf(alphaE[t] - mbuf[d * HGAT + h]);
    alphaE[t] = p;
    atomicAdd(&denom[d * HGAT + h], p);
}

__global__ void edge_scatter(const int* __restrict__ ei, const float* __restrict__ alphaE,
                             const float* __restrict__ XS, float* __restrict__ ACC,
                             int Eo, int Nn) {
    int t = blockIdx.x * 256 + threadIdx.x;     // t = e*64 + c
    int e = t >> 6, c = t & 63;
    if (e >= Eo + Nn) return;
    int s = (e < Eo) ? ei[e] : (e - Eo);
    int d = (e < Eo) ? ei[Eo + e] : (e - Eo);
    const float* xr = XS + (size_t)s * HCG + c;
    float* ar = ACC + (size_t)d * HCG + c;
    float p0 = alphaE[e * 4], p1 = alphaE[e * 4 + 1],
          p2 = alphaE[e * 4 + 2], p3 = alphaE[e * 4 + 3];
    atomicAdd(ar,       p0 * xr[0]);
    atomicAdd(ar + 64,  p1 * xr[64]);
    atomicAdd(ar + 128, p2 * xr[128]);
    atomicAdd(ar + 192, p3 * xr[192]);
}

// ---------------- node finalize / batchnorm ----------------

// normalize by denom, add bias, accumulate BN stats (chunk of 128 nodes / block)
__global__ __launch_bounds__(256)
void node_fin1(float* __restrict__ ACC, const float* __restrict__ denom,
               const float* __restrict__ b1, float* __restrict__ bnS,
               float* __restrict__ bnQ, int Nn) {
    int j = threadIdx.x;                // channel 0..255
    int h = j >> 6;
    int n0 = blockIdx.x * 128;
    float s = 0.f, q = 0.f;
    for (int i = 0; i < 128; ++i) {
        int n = n0 + i;
        if (n >= Nn) break;
        size_t idx = (size_t)n * HCG + j;
        float v = ACC[idx] / (denom[n * HGAT + h] + 1e-16f) + b1[j];
        ACC[idx] = v;
        s += v; q += v * v;
    }
    atomicAdd(&bnS[j], s);
    atomicAdd(&bnQ[j], q);
}

__global__ void bn1_apply(const float* __restrict__ ACC, const float* __restrict__ bnS,
                          const float* __restrict__ bnQ, const float* __restrict__ g,
                          const float* __restrict__ be, _Float16* __restrict__ H16, int Nn) {
    int t = blockIdx.x * 256 + threadIdx.x;
    if (t >= Nn * HCG) return;
    int j = t & 255;
    float invN = 1.f / (float)Nn;
    float mu = bnS[j] * invN;
    float var = bnQ[j] * invN - mu * mu;
    float v = (ACC[t] - mu) * rsqrtf(var + 1e-5f) * g[j] + be[j];
    v = (v > 0.f) ? v : expm1f(v);              // ELU
    H16[t] = (_Float16)v;
}

__global__ __launch_bounds__(64)
void node_fin2(const float* __restrict__ ACC, const float* __restrict__ denom,
               const float* __restrict__ b2, float* __restrict__ OUT2,
               float* __restrict__ bnS, float* __restrict__ bnQ, int Nn) {
    int c = threadIdx.x;                // channel 0..63
    int n0 = blockIdx.x * 256;
    float s = 0.f, q = 0.f;
    for (int i = 0; i < 256; ++i) {
        int n = n0 + i;
        if (n >= Nn) break;
        float i0 = 1.f / (denom[n * 4 + 0] + 1e-16f);
        float i1 = 1.f / (denom[n * 4 + 1] + 1e-16f);
        float i2 = 1.f / (denom[n * 4 + 2] + 1e-16f);
        float i3 = 1.f / (denom[n * 4 + 3] + 1e-16f);
        const float* a = ACC + (size_t)n * HCG + c;
        float v = 0.25f * (a[0] * i0 + a[64] * i1 + a[128] * i2 + a[192] * i3) + b2[c];
        OUT2[(size_t)n * CGAT + c] = v;
        s += v; q += v * v;
    }
    atomicAdd(&bnS[c], s);
    atomicAdd(&bnQ[c], q);
}

// ---------------- head: BN2 + ELU + residual + fc1/relu + fc2/softplus ----------------

__global__ __launch_bounds__(256)
void final_head(const float* __restrict__ OUT2, const float* __restrict__ bnS,
                const float* __restrict__ bnQ, const float* __restrict__ g2,
                const float* __restrict__ be2, const float* __restrict__ RES,
                const float* __restrict__ fc1_W, const float* __restrict__ fc1_b,
                const float* __restrict__ fc2_W, const float* __restrict__ fc2_b,
                float* __restrict__ out, int Nn) {
    __shared__ float vsh[8][64];
    int wave = threadIdx.x >> 5, lane = threadIdx.x & 31;
    int n = blockIdx.x * 8 + wave;
    bool active = (n < Nn);
    if (active) {
        float invN = 1.f / (float)Nn;
        for (int c = lane; c < 64; c += 32) {
            float mu = bnS[c] * invN;
            float var = bnQ[c] * invN - mu * mu;
            float v = (OUT2[(size_t)n * CGAT + c] - mu) * rsqrtf(var + 1e-5f) * g2[c] + be2[c];
            v = (v > 0.f) ? v : expm1f(v);      // ELU
            v += RES[(size_t)n * CGAT + c];     // residual (skip_b added in GEMM)
            vsh[wave][c] = v;
        }
    }
    __syncthreads();
    float r = 0.f;
    if (active) {
        for (int c = 0; c < 64; ++c)
            r += vsh[wave][c] * fc1_W[c * 32 + lane];
        r += fc1_b[lane];
        r = fmaxf(r, 0.f);
        r *= fc2_W[lane];
    }
#pragma unroll
    for (int off = 16; off >= 1; off >>= 1)
        r += __shfl_down(r, off, 32);
    if (active && lane == 0) {
        float z = r + fc2_b[0];
        out[n] = fmaxf(z, 0.f) + log1pf(__expf(-fabsf(z)));   // stable softplus
    }
}

// ---------------- host launch ----------------

static inline int cdiv(long long a, long long b) { return (int)((a + b - 1) / b); }

extern "C" void kernel_launch(void* const* d_in, const int* in_sizes, int n_in,
                              void* d_out, int out_size, void* d_ws, size_t ws_size,
                              hipStream_t stream) {
    const int Nn = in_sizes[0] / 128;          // 50000
    const int Eo = in_sizes[1] / 2;            // 800000
    const int Et = Eo + Nn;

    const float* x      = (const float*)d_in[0];
    const int*   ei     = (const int*)d_in[1];
    const float* eattr  = (const float*)d_in[2];
    const float* W1     = (const float*)d_in[3];
    const float* asrc1  = (const float*)d_in[4];
    const float* adst1  = (const float*)d_in[5];
    const float* We1    = (const float*)d_in[6];
    const float* ae1    = (const float*)d_in[7];
    const float* b1     = (const float*)d_in[8];
    const float* g1     = (const float*)d_in[9];
    const float* be1    = (const float*)d_in[10];
    const float* W2     = (const float*)d_in[11];
    const float* asrc2  = (const float*)d_in[12];
    const float* adst2  = (const float*)d_in[13];
    const float* We2    = (const float*)d_in[14];
    const float* ae2    = (const float*)d_in[15];
    const float* b2     = (const float*)d_in[16];
    const float* g2     = (const float*)d_in[17];
    const float* be2    = (const float*)d_in[18];
    const float* skipW  = (const float*)d_in[19];
    const float* skipb  = (const float*)d_in[20];
    const float* fc1W   = (const float*)d_in[21];
    const float* fc1b   = (const float*)d_in[22];
    const float* fc2W   = (const float*)d_in[23];
    const float* fc2b   = (const float*)d_in[24];
    float* out = (float*)d_out;

    // workspace carve-up (256B aligned)
    char* ws = (char*)d_ws;
    size_t off = 0;
    auto carve = [&](size_t bytes) { size_t o = off; off += (bytes + 255) & ~(size_t)255; return (void*)(ws + o); };
    _Float16* xh16 = (_Float16*)carve((size_t)Nn * 256 * 2);   // x_f16 first, h1_f16 later
    float* XS    = (float*)carve((size_t)Nn * 256 * 4);
    float* ACC   = (float*)carve((size_t)Nn * 256 * 4);
    float* RES   = (float*)carve((size_t)Nn * 64 * 4);
    float* OUT2  = (float*)carve((size_t)Nn * 64 * 4);
    float* AE    = (float*)carve((size_t)Et * 4 * 4);
    float* meanA = (float*)carve((size_t)Nn * 4 * 4);
    float* sums  = (float*)carve((size_t)Nn * 4 * 4);
    float* cnt   = (float*)carve((size_t)Nn * 4);
    float* nAS   = (float*)carve((size_t)Nn * 4 * 4);
    float* nAD   = (float*)carve((size_t)Nn * 4 * 4);
    float* mbuf  = (float*)carve((size_t)Nn * 4 * 4);
    float* denom = (float*)carve((size_t)Nn * 4 * 4);
    float* bnS   = (float*)carve(256 * 4);
    float* bnQ   = (float*)carve(256 * 4);
    float* bn2S  = (float*)carve(64 * 4);
    float* bn2Q  = (float*)carve(64 * 4);
    float* Me1   = (float*)carve(16 * 4);
    float* Me2   = (float*)carve(16 * 4);
    _Float16* W1p = (_Float16*)carve((size_t)128 * 256 * 2);
    _Float16* W2p = (_Float16*)carve((size_t)256 * 256 * 2);
    _Float16* Wsp = (_Float16*)carve((size_t)128 * 64 * 2);

    // ---- self-loop mean attr ----
    fill_f32<<<cdiv(Nn * 4, 256), 256, 0, stream>>>(sums, 0.f, Nn * 4);
    fill_f32<<<cdiv(Nn, 256), 256, 0, stream>>>(cnt, 0.f, Nn);
    edge_mean_acc<<<cdiv(Eo, 256), 256, 0, stream>>>(ei, eattr, sums, cnt, Eo);
    compute_mean<<<cdiv(Nn * 4, 256), 256, 0, stream>>>(sums, cnt, meanA, Nn);

    // ---- f16 conversions + weight packing ----
    conv_f16<<<cdiv((long long)Nn * 128, 256), 256, 0, stream>>>(x, xh16, Nn * 128);
    pack_b<<<cdiv(16 * 4 * 256, 256), 256, 0, stream>>>(W1, W1p, 128, 256);
    pack_b<<<cdiv(16 * 8 * 256, 256), 256, 0, stream>>>(W2, W2p, 256, 256);
    pack_b<<<cdiv(4 * 4 * 256, 256), 256, 0, stream>>>(skipW, Wsp, 128, 64);
    compute_me<<<1, 32, 0, stream>>>(We1, ae1, Me1);
    compute_me<<<1, 32, 0, stream>>>(We2, ae2, Me2);

    // ---- residual GEMM (bias fused) + layer-1 feature GEMM ----
    gemm_wmma_f16<<<cdiv((Nn / 16) * 4, 4), 128, 0, stream>>>(xh16, Wsp, RES, skipb, Nn, 128, 64);
    gemm_wmma_f16<<<cdiv((Nn / 16) * 16, 4), 128, 0, stream>>>(xh16, W1p, XS, nullptr, Nn, 128, 256);

    // ---- layer 1 attention ----
    alpha_node<<<cdiv(Nn * 4, 256), 256, 0, stream>>>(XS, asrc1, adst1, nAS, nAD, Nn);
    fill_f32<<<cdiv(Nn * 4, 256), 256, 0, stream>>>(mbuf, -1e30f, Nn * 4);
    fill_f32<<<cdiv(Nn * 4, 256), 256, 0, stream>>>(denom, 0.f, Nn * 4);
    edge_alpha_max<<<cdiv((long long)Et * 4, 256), 256, 0, stream>>>(ei, eattr, meanA, Me1, nAS, nAD, AE, mbuf, Eo, Nn);
    edge_exp_denom<<<cdiv((long long)Et * 4, 256), 256, 0, stream>>>(ei, AE, mbuf, denom, Eo, Nn);
    fill_f32<<<cdiv((long long)Nn * 256, 256), 256, 0, stream>>>(ACC, 0.f, Nn * 256);
    edge_scatter<<<cdiv((long long)Et * 64, 256), 256, 0, stream>>>(ei, AE, XS, ACC, Eo, Nn);
    fill_f32<<<1, 256, 0, stream>>>(bnS, 0.f, 256);
    fill_f32<<<1, 256, 0, stream>>>(bnQ, 0.f, 256);
    node_fin1<<<cdiv(Nn, 128), 256, 0, stream>>>(ACC, denom, b1, bnS, bnQ, Nn);
    bn1_apply<<<cdiv((long long)Nn * 256, 256), 256, 0, stream>>>(ACC, bnS, bnQ, g1, be1, xh16, Nn);

    // ---- layer 2 ----
    gemm_wmma_f16<<<cdiv((Nn / 16) * 16, 4), 128, 0, stream>>>(xh16, W2p, XS, nullptr, Nn, 256, 256);
    alpha_node<<<cdiv(Nn * 4, 256), 256, 0, stream>>>(XS, asrc2, adst2, nAS, nAD, Nn);
    fill_f32<<<cdiv(Nn * 4, 256), 256, 0, stream>>>(mbuf, -1e30f, Nn * 4);
    fill_f32<<<cdiv(Nn * 4, 256), 256, 0, stream>>>(denom, 0.f, Nn * 4);
    edge_alpha_max<<<cdiv((long long)Et * 4, 256), 256, 0, stream>>>(ei, eattr, meanA, Me2, nAS, nAD, AE, mbuf, Eo, Nn);
    edge_exp_denom<<<cdiv((long long)Et * 4, 256), 256, 0, stream>>>(ei, AE, mbuf, denom, Eo, Nn);
    fill_f32<<<cdiv((long long)Nn * 256, 256), 256, 0, stream>>>(ACC, 0.f, Nn * 256);
    edge_scatter<<<cdiv((long long)Et * 64, 256), 256, 0, stream>>>(ei, AE, XS, ACC, Eo, Nn);
    fill_f32<<<1, 64, 0, stream>>>(bn2S, 0.f, 64);
    fill_f32<<<1, 64, 0, stream>>>(bn2Q, 0.f, 64);
    node_fin2<<<cdiv(Nn, 256), 64, 0, stream>>>(ACC, denom, b2, OUT2, bn2S, bn2Q, Nn);

    // ---- head ----
    final_head<<<cdiv(Nn, 8), 256, 0, stream>>>(OUT2, bn2S, bn2Q, g2, be2, RES,
                                                fc1W, fc1b, fc2W, fc2b, out, Nn);
    (void)n_in; (void)out_size; (void)ws_size;
}
